// LPVDynamics_85882166050860
// MI455X (gfx1250) — matrix operand order repaired and checked by reading, not compiled, not run on hardware
//
#include <hip/hip_runtime.h>

typedef __attribute__((ext_vector_type(16))) _Float16 v16h;
typedef __attribute__((ext_vector_type(8)))  _Float16 v8h;
typedef __attribute__((ext_vector_type(8)))  float    v8f;
typedef __attribute__((ext_vector_type(4)))  float    v4f;

#define BATCH   262144
#define ZD      128
#define RD      16
#define GD      16
#define HID     256
#define NTILES  (BATCH / 16)

#define GAMMA   0.1f
#define CAPV    2.2627416997969522f   // 0.5 * 0.1 * sqrt(128*16)
#define LN10F   2.302585092994046f
#define DTLMIN  (-3.0f)
#define DTRANGE 11.0f
#define DTLCAP  2.7781512503836436f   // min(8, log10(60/0.1))

// LDS layout (units: _Float16)
#define W1_OFF   0
#define W1_SZ    (16 * 32 * 16)         // 16 n-frags, full width (upper K half pre-zeroed)
#define W2_OFF   (W1_OFF + W1_SZ)
#define W2_SZ    (128 * 32 * 16)        // 8 kc x 16 nb frags
#define W3_OFF   (W2_OFF + W2_SZ)
#define W3_SZ    (8 * 32 * 16)          // 8 kc frags
#define UP_OFF   (W3_OFF + W3_SZ)
#define UP_SZ    (4 * 32 * 16)          // z@U : 4 kc frags
#define UT_OFF   (UP_OFF + UP_SZ)
#define UT_SZ    (8 * 32 * 16)          // w@U^T : 8 nz frags
#define HB_OFF   (UT_OFF + UT_SZ)
#define HB_SZ    (4 * 16 * HID)         // 4 waves x 16x256 h-scratch
#define WB_OFF   (HB_OFF + HB_SZ)
#define WB_SZ    (4 * 16 * 32)          // 4 waves x 16x32 coeff tile (upper K half stays 0)
#define SMEM_F16 (WB_OFF + WB_SZ)
#define SMEM_BYTES (SMEM_F16 * 2 + 16 * 4 /* colnorm2 */)

__device__ __forceinline__ float nan0(float x) {
    return __builtin_isfinite(x) ? x : 0.0f;
}
// fast sigmoid via hardware exp/rcp: v_rcp_f32(1 + v_exp_f32(-x))
__device__ __forceinline__ float fast_sigmoid(float x) {
    return __builtin_amdgcn_rcpf(1.0f + __expf(-x));
}
__device__ __forceinline__ void wave_sync() {
    __builtin_amdgcn_wave_barrier();   // stop compiler reordering around per-wave LDS RAW
}
__device__ __forceinline__ v16h lds_frag16(const _Float16* p) {   // 32B fragment as 2x16B
    v8h lo = *(const v8h*)p;
    v8h hi = *(const v8h*)(p + 8);
    v16h r;
#pragma unroll
    for (int j = 0; j < 8; ++j) { r[j] = lo[j]; r[8 + j] = hi[j]; }
    return r;
}
#define WMMA(A, B, C) __builtin_amdgcn_wmma_f32_16x16x32_f16(false, (A), false, (B), (short)0, (C), false, false)

// 16-bit fragment K index for element j of a lane (ISA 7.12.2 layout)
__device__ __forceinline__ int frag_k(int kb, int j) {
    return (j < 8) ? (kb + j) : (16 + kb + (j - 8));
}

__global__ __launch_bounds__(128) void lpv_dyn_kernel(
    const float* __restrict__ pt, const float* __restrict__ z,
    const float* __restrict__ dtn, const float* __restrict__ U,
    const float* __restrict__ W1, const float* __restrict__ b1,
    const float* __restrict__ W2, const float* __restrict__ b2,
    const float* __restrict__ W3, const float* __restrict__ b3,
    float* __restrict__ out)
{
    extern __shared__ _Float16 smem[];
    _Float16* w1p = smem + W1_OFF;
    _Float16* w2p = smem + W2_OFF;
    _Float16* w3p = smem + W3_OFF;
    _Float16* up  = smem + UP_OFF;
    _Float16* utp = smem + UT_OFF;
    _Float16* hbA = smem + HB_OFF;
    _Float16* wbA = smem + WB_OFF;
    float*    cn2 = (float*)(smem + SMEM_F16);

    const int tid  = threadIdx.x;
    const int lane = tid & 31;
    const int wave = tid >> 5;
    const int nIdx = lane & 15;            // N / M position inside fragments
    const int kb   = (lane < 16) ? 0 : 8;  // half-wave K base

    // ---------------- pack weights (f32 global -> f16 WMMA B-fragments in LDS) ----------------
    for (int s = tid; s < 16 * 32; s += 128) {       // W1: K=16 (pad 32, zeros baked in)
        int nb = s >> 5, l = s & 31;
        int n = nb * 16 + (l & 15), kbp = (l < 16) ? 0 : 8;
#pragma unroll
        for (int j = 0; j < 16; ++j) {
            int k = frag_k(kbp, j);
            w1p[s * 16 + j] = (k < GD) ? (_Float16)W1[k * HID + n] : (_Float16)0.0f;
        }
    }
    for (int s = tid; s < 128 * 32; s += 128) {      // W2: 8 kc x 16 nb
        int f = s >> 5, l = s & 31;
        int kc = f >> 4, nb = f & 15;
        int n = nb * 16 + (l & 15), kbp = (l < 16) ? 0 : 8;
#pragma unroll
        for (int j = 0; j < 16; ++j)
            w2p[s * 16 + j] = (_Float16)W2[(kc * 32 + frag_k(kbp, j)) * HID + n];
    }
    for (int s = tid; s < 8 * 32; s += 128) {        // W3: 8 kc, N=16
        int kc = s >> 5, l = s & 31;
        int n = l & 15, kbp = (l < 16) ? 0 : 8;
#pragma unroll
        for (int j = 0; j < 16; ++j)
            w3p[s * 16 + j] = (_Float16)W3[(kc * 32 + frag_k(kbp, j)) * RD + n];
    }
    for (int s = tid; s < 4 * 32; s += 128) {        // U as B for z@U : K=Z(128), N=R
        int kc = s >> 5, l = s & 31;
        int n = l & 15, kbp = (l < 16) ? 0 : 8;
#pragma unroll
        for (int j = 0; j < 16; ++j)
            up[s * 16 + j] = (_Float16)nan0(U[(kc * 32 + frag_k(kbp, j)) * RD + n]);
    }
    for (int s = tid; s < 8 * 32; s += 128) {        // U^T as B for w@U^T : K=R (pad 32), N=Z
        int nb = s >> 5, l = s & 31;
        int nz = nb * 16 + (l & 15), kbp = (l < 16) ? 0 : 8;
#pragma unroll
        for (int j = 0; j < 16; ++j) {
            int kr = frag_k(kbp, j);
            utp[s * 16 + j] = (kr < RD) ? (_Float16)nan0(U[nz * RD + kr]) : (_Float16)0.0f;
        }
    }
    for (int i = tid; i < WB_SZ; i += 128)           // coeff tile scratch: zero once;
        wbA[i] = (_Float16)0.0f;                     // per tile only K<16 is rewritten
    if (tid < 16) {                                   // column norms of U
        float acc = 0.0f;
        for (int zz = 0; zz < ZD; ++zz) { float u = nan0(U[zz * RD + tid]); acc += u * u; }
        cn2[tid] = acc;
    }
    __syncthreads();

    // per-lane hoisted constants
    float b1v[16], b2v[16];
#pragma unroll
    for (int nb = 0; nb < 16; ++nb) {
        b1v[nb] = b1[nb * 16 + nIdx];
        b2v[nb] = b2[nb * 16 + nIdx];
    }
    const float b3v = b3[nIdx];
    const float cnv = cn2[nIdx];

    _Float16* hb = hbA + wave * 16 * HID;
    _Float16* wb = wbA + wave * 16 * 32;
    const int gwave  = blockIdx.x * 4 + wave;
    const int nwaves = gridDim.x * 4;

    for (int t = gwave; t < NTILES; t += nwaves) {
        const int m0   = t * 16;
        const int mrow = m0 + nIdx;   // A-fragment row this lane serves

        // prefetch next tile's streaming inputs
        if (t + nwaves < NTILES) {
            __builtin_prefetch(z  + (size_t)(m0 + nwaves * 16) * ZD + lane * 16, 0, 0);
            __builtin_prefetch(pt + (size_t)(m0 + nwaves * 16) * GD + lane * 8, 0, 0);
        }

        // ---- A fragment from pt (K=16, zero-pad to 32) ----
        v16h apt;
        {
            const float* p = pt + mrow * GD + kb;
            v4f p0 = *(const v4f*)(p);
            v4f p1 = *(const v4f*)(p + 4);
#pragma unroll
            for (int j = 0; j < 4; ++j) {
                apt[j]      = (_Float16)nan0(p0[j]);
                apt[4 + j]  = (_Float16)nan0(p1[j]);
                apt[8 + j]  = (_Float16)0.0f;
                apt[12 + j] = (_Float16)0.0f;
            }
        }

        // ---- layer 1: h1 = silu(pt@W1 + b1) -> LDS ----
#pragma unroll
        for (int nb = 0; nb < 16; ++nb) {
            v16h bf = lds_frag16(w1p + (nb * 32 + lane) * 16);
            v8f c = {};
            c = WMMA(apt, bf, c);
#pragma unroll
            for (int v = 0; v < 8; ++v) {
                float h = c[v] + b1v[nb];
                h = h * fast_sigmoid(h);
                hb[(v + kb) * HID + nb * 16 + nIdx] = (_Float16)h;
            }
        }
        wave_sync();

        // ---- layer 2 A fragments (LDS transpose read) ----
        v16h a2[8];
#pragma unroll
        for (int kc = 0; kc < 8; ++kc)
            a2[kc] = __builtin_shufflevector(*(const v8h*)(hb + nIdx * HID + kc * 32 + kb), *(const v8h*)(hb + nIdx * HID + kc * 32 + kb + 16), 0, 1, 2, 3, 4, 5, 6, 7, 8, 9, 10, 11, 12, 13, 14, 15);   // [0..7]=kb.., [8..15]=16+kb..
        wave_sync();

        // ---- layer 2: h2 = silu(h1@W2 + b2) -> LDS (overwrite) ----
#pragma unroll
        for (int nb = 0; nb < 16; ++nb) {
            v8f c = {};
#pragma unroll
            for (int kc = 0; kc < 8; ++kc) {
                v16h bf = lds_frag16(w2p + ((kc * 16 + nb) * 32 + lane) * 16);
                c = WMMA(a2[kc], bf, c);
            }
#pragma unroll
            for (int v = 0; v < 8; ++v) {
                float h = c[v] + b2v[nb];
                h = h * fast_sigmoid(h);
                hb[(v + kb) * HID + nb * 16 + nIdx] = (_Float16)h;
            }
        }
        wave_sync();

        // ---- layer 3: logits = h2@W3 + b3 ----
        v8f c3 = {};
#pragma unroll
        for (int kc = 0; kc < 8; ++kc) {
            v16h a3 = __builtin_shufflevector(*(const v8h*)(hb + nIdx * HID + kc * 32 + kb), *(const v8h*)(hb + nIdx * HID + kc * 32 + kb + 16), 0, 1, 2, 3, 4, 5, 6, 7, 8, 9, 10, 11, 12, 13, 14, 15);
            v16h bf = lds_frag16(w3p + (kc * 32 + lane) * 16);
            c3 = WMMA(a3, bf, c3);
        }
        wave_sync();

        // ---- s = 2*sigmoid, Frobenius cap (reduce over 16 R-lanes), dt denorm ----
        float sv[8], lamg[8], coef[8];
#pragma unroll
        for (int v = 0; v < 8; ++v)
            sv[v] = 2.0f * fast_sigmoid(c3[v] + b3v);
#pragma unroll
        for (int v = 0; v < 8; ++v) {
            float x = sv[v] * sv[v] * cnv;
            x += __shfl_xor(x, 1, 32);
            x += __shfl_xor(x, 2, 32);
            x += __shfl_xor(x, 4, 32);
            x += __shfl_xor(x, 8, 32);
            // cap / max(sqrt(x),1e-9) == cap * rsqrt(max(x,1e-18)) ; one v_rsq_f32
            float scale = fminf(CAPV * __builtin_amdgcn_rsqf(fmaxf(x, 1e-18f)), 1.0f);
            sv[v] *= scale;
        }
#pragma unroll
        for (int v = 0; v < 8; ++v) {
            float d = dtn[m0 + v + kb];
            d = (d != d) ? 0.0f : fminf(fmaxf(d, 0.0f), 1.0f);
            float ld = fminf(DTLMIN + d * DTRANGE, DTLCAP);
            float dp = fmaxf(__expf(ld * LN10F), 1e-30f);
            lamg[v] = __expf(-GAMMA * dp);
            coef[v] = __expf(-(sv[v] * sv[v]) * dp) * lamg[v] - lamg[v];   // lam_para - lam_g
        }

        // ---- uTz = z @ U  (K = 128 over 4 chunks) ----
        v8f cu = {};
#pragma unroll
        for (int kc = 0; kc < 4; ++kc) {
            const float* zp = z + mrow * ZD + kc * 32 + kb;
            v4f z0 = *(const v4f*)(zp);
            v4f z1 = *(const v4f*)(zp + 4);
            v4f z2 = *(const v4f*)(zp + 16);
            v4f z3 = *(const v4f*)(zp + 20);
            v16h az;
#pragma unroll
            for (int j = 0; j < 4; ++j) {
                az[j]      = (_Float16)nan0(z0[j]);
                az[4 + j]  = (_Float16)nan0(z1[j]);
                az[8 + j]  = (_Float16)nan0(z2[j]);
                az[12 + j] = (_Float16)nan0(z3[j]);
            }
            v16h bf = lds_frag16(up + (kc * 32 + lane) * 16);
            cu = WMMA(az, bf, cu);
        }

        // ---- w = (lam_para - lam_g) * uTz ; transpose via pre-zero-padded LDS tile ----
#pragma unroll
        for (int v = 0; v < 8; ++v)
            wb[(v + kb) * 32 + nIdx] = (_Float16)(coef[v] * cu[v]);   // K>=16 half stays 0
        wave_sync();
        v16h aw = __builtin_shufflevector(*(const v8h*)(wb + nIdx * 32 + kb), *(const v8h*)(wb + nIdx * 32 + kb + 16), 0, 1, 2, 3, 4, 5, 6, 7, 8, 9, 10, 11, 12, 13, 14, 15);
        wave_sync();

        // ---- z_next = lam_g * z + w @ U^T ----
#pragma unroll
        for (int nb = 0; nb < 8; ++nb) {
            v16h bf = lds_frag16(utp + (nb * 32 + lane) * 16);
            v8f cz = {};
            cz = WMMA(aw, bf, cz);
#pragma unroll
            for (int v = 0; v < 8; ++v) {
                int row = m0 + v + kb;
                int n   = nb * 16 + nIdx;
                float zv = nan0(z[row * ZD + n]);
                out[row * ZD + n] = nan0(lamg[v] * zv + cz[v]);
            }
        }
    }
}

extern "C" void kernel_launch(void* const* d_in, const int* in_sizes, int n_in,
                              void* d_out, int out_size, void* d_ws, size_t ws_size,
                              hipStream_t stream) {
    (void)in_sizes; (void)n_in; (void)d_ws; (void)ws_size; (void)out_size;
    const float* pt  = (const float*)d_in[0];
    const float* z   = (const float*)d_in[1];
    const float* dtn = (const float*)d_in[2];
    const float* U   = (const float*)d_in[3];
    const float* W1  = (const float*)d_in[4];
    const float* b1  = (const float*)d_in[5];
    const float* W2  = (const float*)d_in[6];
    const float* b2  = (const float*)d_in[7];
    const float* W3  = (const float*)d_in[8];
    const float* b3  = (const float*)d_in[9];
    float* out = (float*)d_out;

    const size_t smem = (size_t)SMEM_BYTES;   // ~205 KB, fits 320 KB/WGP LDS
    lpv_dyn_kernel<<<1024, 128, smem, stream>>>(pt, z, dtn, U, W1, b1, W2, b2, W3, b3, out);
}
